// MemoryMHA_7645041787134
// MI455X (gfx1250) — matrix operand order repaired and checked by reading, not compiled, not run on hardware
//
#include <hip/hip_runtime.h>
#include <hip/hip_bf16.h>

#define DD  768
#define MMEM 16
#define SS  1024
#define BB  16
#define KV  1040   /* S + M */
#define KVP 1056   /* padded to multiple of 32 for WMMA K-dim */
#define SCALE_F 0.03608439182435161f  /* 1/sqrt(768) */

typedef __attribute__((ext_vector_type(16))) __bf16 v16bf;
typedef __attribute__((ext_vector_type(8)))  __bf16 v8bf;
typedef __attribute__((ext_vector_type(8)))  float  v8f;

static __device__ __forceinline__ v8f zero8() {
  v8f z;
#pragma unroll
  for (int i = 0; i < 8; ++i) z[i] = 0.0f;
  return z;
}

// ---------------------------------------------------------------------------
// Generic bf16 WMMA GEMM:  C[m][n] = alpha * sum_k A[m][k] * B[n][k] (+ bias)
// A: row-major [M,K] bf16, B: row-major [N,K] bf16 (i.e. C = A * B^T)
// Block = 256 threads = 8 waves (4 m-waves x 2 n-waves); block tile 256x128;
// wave tile 64x64 = 4x4 WMMA 16x16 tiles. M must be a multiple of 256,
// K a multiple of 32, N a multiple of 16.
// N-edge handling: out-of-range 16-wide tiles *load* from a clamped (valid)
// row so the K loop is completely branch-free (EXEC stays all-ones around
// WMMA); their stores are predicated off instead.
// bias_mode: 0 none, 1 per-column, 2 per-row. out_bf16: 1 bf16 out, 0 f32.
// ---------------------------------------------------------------------------
__global__ __launch_bounds__(256)
void gemm_wmma_kernel(const __bf16* __restrict__ A, long long sA, int lda,
                      const __bf16* __restrict__ B, long long sB, int ldb,
                      void* __restrict__ C, long long sC, int ldc,
                      int Ncols, int Kdim,
                      const float* __restrict__ bias, int bias_mode,
                      float alpha, int out_bf16)
{
  const int z = (int)blockIdx.z;
  const __bf16* Ab = A + (long long)z * sA;
  const __bf16* Bb = B + (long long)z * sB;

  const int lane = (int)(threadIdx.x & 31u);
  const int wave = (int)(threadIdx.x >> 5u);
  const int wm = wave & 3;       // 0..3  -> m offset within block
  const int wn = wave >> 2;      // 0..1  -> n offset within block
  const int r  = lane & 15;
  const int h  = lane >> 4;
  const int mwave = (int)blockIdx.x * 256 + wm * 64;
  const int nwave = (int)blockIdx.y * 128 + wn * 64;

  v8f acc[4][4];
#pragma unroll
  for (int i = 0; i < 4; ++i)
#pragma unroll
    for (int j = 0; j < 4; ++j) acc[i][j] = zero8();

  // Store predicate + clamped load row for N edge tiles (wave-uniform).
  bool nok[4];
  const __bf16* pb0[4];
#pragma unroll
  for (int j = 0; j < 4; ++j) {
    const int nt = nwave + j * 16;
    nok[j] = nt < Ncols;
    const int ntc = nok[j] ? nt : (Ncols - 16);   // clamp to a valid tile
    pb0[j] = Bb + (long long)(ntc + r) * ldb + h * 16;
  }
  const __bf16* pa0[4];
#pragma unroll
  for (int i = 0; i < 4; ++i)
    pa0[i] = Ab + (long long)(mwave + i * 16 + r) * lda + h * 16;

  for (int k0 = 0; k0 < Kdim; k0 += 32) {
    v16bf af[4], bf[4];
    // A fragment (16x32 bf16): lane (m=r, half=h) holds K = h*16..h*16+15
    // -> two contiguous 16-byte loads from row m.
#pragma unroll
    for (int i = 0; i < 4; ++i) {
      const __bf16* pa = pa0[i] + k0;
      union { v16bf v; v8bf p[2]; } u;
      u.p[0] = *(const v8bf*)pa;
      u.p[1] = *(const v8bf*)(pa + 8);
      af[i] = u.v;
    }
    // B fragment (32x16 bf16): lane n=r holds K = h*16..h*16+15 of column n,
    // i.e. one contiguous 32-byte read of row n of our row-major B operand.
#pragma unroll
    for (int j = 0; j < 4; ++j)
      bf[j] = *(const v16bf*)(pb0[j] + k0);

#pragma unroll
    for (int i = 0; i < 4; ++i)
#pragma unroll
      for (int j = 0; j < 4; ++j)
        acc[i][j] = __builtin_amdgcn_wmma_f32_16x16x32_bf16(
            false, af[i], false, bf[j], (short)0, acc[i][j], false, false);
  }

  // C/D layout: element e of v8f in lane L -> row = e + 8*(L>>4), col = L&15
  __bf16* Cb16 = (__bf16*)C + (long long)z * sC;
  float*  Cf32 = (float*)C  + (long long)z * sC;
#pragma unroll
  for (int i = 0; i < 4; ++i) {
#pragma unroll
    for (int j = 0; j < 4; ++j) {
      if (!nok[j]) continue;
      const int col = nwave + j * 16 + r;
      const float bc = (bias_mode == 1) ? bias[col] : 0.0f;
#pragma unroll
      for (int e = 0; e < 8; ++e) {
        const int row = mwave + i * 16 + e + 8 * h;
        float val = acc[i][j][e] * alpha + bc;
        if (bias_mode == 2) val += bias[row];
        const long long o = (long long)row * ldc + col;
        if (out_bf16) Cb16[o] = (__bf16)val;
        else          Cf32[o] = val;
      }
    }
  }
}

// ---------------------------------------------------------------------------
// Row softmax over kv (1040 valid cols, 1056 stored): attn = softmax(sc+mask)
// One 256-thread block per (s, b) row; pad columns written as exact zeros so
// the padded K-dim of the attn @ V GEMM contributes nothing.
// ---------------------------------------------------------------------------
__global__ __launch_bounds__(256)
void softmax_kernel(const float* __restrict__ scores,
                    const float* __restrict__ mask,
                    __bf16* __restrict__ attn)
{
  const int s = (int)blockIdx.x;
  const int b = (int)blockIdx.y;
  const float* row  = scores + ((long long)b * SS + s) * KVP;
  const float* mrow = mask   + (long long)s * KV;
  __bf16* arow      = attn   + ((long long)b * SS + s) * KVP;
  const int t = (int)threadIdx.x;

  float v[5];
  float lmax = -3.4e38f;
#pragma unroll
  for (int i = 0; i < 5; ++i) {
    const int c = t + i * 256;
    float val = -3.4e38f;
    if (c < KV) val = row[c] + mrow[c];
    v[i] = val;
    lmax = fmaxf(lmax, val);
  }
  __shared__ float red[256];
  red[t] = lmax;
  __syncthreads();
  for (int o = 128; o > 0; o >>= 1) {
    if (t < o) red[t] = fmaxf(red[t], red[t + o]);
    __syncthreads();
  }
  const float rmax = red[0];
  __syncthreads();

  float lsum = 0.0f;
#pragma unroll
  for (int i = 0; i < 5; ++i) {
    const int c = t + i * 256;
    float e = 0.0f;
    if (c < KV) e = __expf(v[i] - rmax);
    v[i] = e;
    lsum += e;
  }
  red[t] = lsum;
  __syncthreads();
  for (int o = 128; o > 0; o >>= 1) {
    if (t < o) red[t] += red[t + o];
    __syncthreads();
  }
  const float inv = 1.0f / red[0];
#pragma unroll
  for (int i = 0; i < 5; ++i) {
    const int c = t + i * 256;
    if (c < KVP) arow[c] = (__bf16)(v[i] * inv);  // pads (>=1040) get 0
  }
}

// Build xcat bf16 [B][KV][D]: rows 0..S-1 from x, rows S.. from memory (bcast)
__global__ __launch_bounds__(256)
void concat_cast_kernel(const float* __restrict__ x, const float* __restrict__ mem,
                        __bf16* __restrict__ xcat)
{
  const long long idx = (long long)blockIdx.x * 256 + threadIdx.x;
  const long long total = (long long)BB * KV * DD;
  if (idx >= total) return;
  const int d = (int)(idx % DD);
  const long long rb = idx / DD;
  const int row = (int)(rb % KV);
  const int b = (int)(rb / KV);
  float val;
  if (row < SS) val = x[((long long)b * SS + row) * DD + d];
  else          val = mem[(long long)(row - SS) * DD + d];
  xcat[idx] = (__bf16)val;
}

__global__ __launch_bounds__(256)
void cast_bf16_kernel(const float* __restrict__ in, __bf16* __restrict__ out, long long n)
{
  const long long idx = (long long)blockIdx.x * 256 + threadIdx.x;
  if (idx < n) out[idx] = (__bf16)in[idx];
}

// Zero the 16 padded columns (1040..1055) of V^T so pad reads are clean
__global__ __launch_bounds__(256)
void vt_pad_kernel(__bf16* __restrict__ vt)
{
  const long long idx = (long long)blockIdx.x * 256 + threadIdx.x;
  const long long total = (long long)BB * DD * (KVP - KV);
  if (idx >= total) return;
  const int c = (int)(idx % (KVP - KV));
  const long long eb = idx / (KVP - KV);
  const int e = (int)(eb % DD);
  const int b = (int)(eb / DD);
  vt[((long long)b * DD + e) * KVP + KV + c] = (__bf16)0.0f;
}

extern "C" void kernel_launch(void* const* d_in, const int* in_sizes, int n_in,
                              void* d_out, int out_size, void* d_ws, size_t ws_size,
                              hipStream_t stream)
{
  const float* x    = (const float*)d_in[0];
  const float* mask = (const float*)d_in[1];
  const float* mem  = (const float*)d_in[2];
  const float* wq   = (const float*)d_in[3];
  const float* bq   = (const float*)d_in[4];
  const float* wk   = (const float*)d_in[5];
  const float* bk   = (const float*)d_in[6];
  const float* wv   = (const float*)d_in[7];
  const float* bv   = (const float*)d_in[8];
  const float* wo   = (const float*)d_in[9];
  const float* bo   = (const float*)d_in[10];
  float* out = (float*)d_out;

  char* ws = (char*)d_ws;
  size_t off = 0;
  auto alloc = [&](size_t bytes) -> void* {
    void* p = ws + off;
    off = (off + bytes + 255) & ~(size_t)255;
    return p;
  };

  __bf16* xcat = (__bf16*)alloc((size_t)BB * KV * DD * 2);   // [B][KV][D]
  __bf16* wqb  = (__bf16*)alloc((size_t)DD * DD * 2);
  __bf16* wkb  = (__bf16*)alloc((size_t)DD * DD * 2);
  __bf16* wvb  = (__bf16*)alloc((size_t)DD * DD * 2);
  __bf16* wob  = (__bf16*)alloc((size_t)DD * DD * 2);
  __bf16* qb   = (__bf16*)alloc((size_t)BB * SS * DD * 2);   // [B][S][D]
  __bf16* kc   = (__bf16*)alloc((size_t)BB * KV * DD * 2);   // [B][KV][D]
  __bf16* vt   = (__bf16*)alloc((size_t)BB * DD * KVP * 2);  // [B][D][KVP]  (V^T)
  float*  sc   = (float*) alloc((size_t)BB * SS * KVP * 4);  // [B][S][KVP]
  __bf16* attn = (__bf16*)alloc((size_t)BB * SS * KVP * 2);  // [B][S][KVP]
  __bf16* head = (__bf16*)alloc((size_t)BB * SS * DD * 2);   // [B][S][D]
  (void)ws_size; (void)in_sizes; (void)n_in; (void)out_size;

  // 1) inputs -> bf16
  {
    const long long total = (long long)BB * KV * DD;
    concat_cast_kernel<<<dim3((unsigned)((total + 255) / 256)), 256, 0, stream>>>(x, mem, xcat);
    const long long nw = (long long)DD * DD;
    const unsigned wb = (unsigned)((nw + 255) / 256);
    cast_bf16_kernel<<<wb, 256, 0, stream>>>(wq, wqb, nw);
    cast_bf16_kernel<<<wb, 256, 0, stream>>>(wk, wkb, nw);
    cast_bf16_kernel<<<wb, 256, 0, stream>>>(wv, wvb, nw);
    cast_bf16_kernel<<<wb, 256, 0, stream>>>(wo, wob, nw);
    const long long np = (long long)BB * DD * (KVP - KV);
    vt_pad_kernel<<<(unsigned)((np + 255) / 256), 256, 0, stream>>>(vt);
  }

  // 2) Q = X * Wq^T + bq            (per batch: M=1024, N=768, K=768)
  gemm_wmma_kernel<<<dim3(SS / 256, DD / 128, BB), 256, 0, stream>>>(
      xcat, (long long)KV * DD, DD,       // A = xcat rows 0..S-1 of each batch
      wqb, 0, DD,
      qb, (long long)SS * DD, DD,
      DD, DD, bq, 1, 1.0f, 1);

  // 3) K = XCat * Wk^T + bk         (flattened: M=16640, N=768, K=768)
  gemm_wmma_kernel<<<dim3((BB * KV) / 256, DD / 128, 1), 256, 0, stream>>>(
      xcat, 0, DD,
      wkb, 0, DD,
      kc, 0, DD,
      DD, DD, bk, 1, 1.0f, 1);

  // 4) V^T = Wv * XCat^T + bv(row)  (per batch: M=768, N=1040, K=768)
  gemm_wmma_kernel<<<dim3(DD / 256, (KVP + 127) / 128, BB), 256, 0, stream>>>(
      wvb, 0, DD,
      xcat, (long long)KV * DD, DD,
      vt, (long long)DD * KVP, KVP,
      KV, DD, bv, 2, 1.0f, 1);

  // 5) scores = scale * Q * K^T     (per batch: M=1024, N=1040, K=768), f32
  gemm_wmma_kernel<<<dim3(SS / 256, (KVP + 127) / 128, BB), 256, 0, stream>>>(
      qb, (long long)SS * DD, DD,
      kc, (long long)KV * DD, DD,
      sc, (long long)SS * KVP, KVP,
      KV, DD, nullptr, 0, SCALE_F, 0);

  // 6) attn = softmax(scores + mask), bf16, zero-padded to KVP
  softmax_kernel<<<dim3(SS, BB), 256, 0, stream>>>(sc, mask, attn);

  // 7) head = attn * (V^T)^T        (per batch: M=1024, N=768, K=1056)
  gemm_wmma_kernel<<<dim3(SS / 256, DD / 128, BB), 256, 0, stream>>>(
      attn, (long long)SS * KVP, KVP,
      vt, (long long)DD * KVP, KVP,
      head, (long long)SS * DD, DD,
      DD, KVP, nullptr, 0, 1.0f, 1);

  // 8) out = head * Wo^T + bo       (flattened: M=16384, N=768, K=768), f32
  gemm_wmma_kernel<<<dim3((BB * SS) / 256, DD / 128, 1), 256, 0, stream>>>(
      head, 0, DD,
      wob, 0, DD,
      out, 0, DD,
      DD, DD, bo, 1, 1.0f, 0);
}
